// BaselineModel_10333691314797
// MI455X (gfx1250) — matrix operand order repaired and checked by reading, not compiled, not run on hardware
//
#include <hip/hip_runtime.h>
#include <hip/hip_bf16.h>
#include <math.h>

// ---------------------------------------------------------------------------
// Model constants (reference: B=2, L=2048, D=256, H=8, HD=32, NB=2)
// ---------------------------------------------------------------------------
#define BATCH 2
#define SEQL  2048
#define DIM   256
#define HEADS 8
#define HDIM  32
#define NBLK  2
#define ROWS  (BATCH * SEQL)          // 4096
#define PROJC (4 * DIM)               // 1024

typedef __attribute__((ext_vector_type(16))) __bf16 v16bf;
typedef __attribute__((ext_vector_type(8)))  float  v8f;

// One v_rcp_f32 instead of the IEEE divide expansion.
static __device__ __forceinline__ float sigm(float x) {
    return __builtin_amdgcn_rcpf(1.0f + __expf(-x));
}

// ---------------------------------------------------------------------------
// Embedding gather: item/user tables selected by mask (0 -> row 0 of table)
// ---------------------------------------------------------------------------
__global__ void gather_k(const int* __restrict__ log_seqs,
                         const int* __restrict__ mask,
                         const float* __restrict__ item_table,
                         const float* __restrict__ user_table,
                         float* __restrict__ item_emb,
                         float* __restrict__ user_emb) {
    int i = blockIdx.x * blockDim.x + threadIdx.x;   // ROWS*DIM threads
    int row = i >> 8;
    int d   = i & 255;
    int ls  = log_seqs[row];
    int m   = mask[row];
    int ii  = (m == 1) ? ls : 0;
    int ui  = (m == 2) ? ls : 0;
    item_emb[i] = item_table[(size_t)ii * DIM + d];
    user_emb[i] = user_table[(size_t)ui * DIM + d];
}

// ---------------------------------------------------------------------------
// bf16-WMMA GEMM: out[ROWS,N] = act(A[ROWS,K] * W[K,N] + bias)
// block = 256 threads (8 waves); wave w -> 16x16 tile (m0 = by*128 + w*16).
// All 8 waves share one n0, so the 32x16 W tile is staged once per k-step in
// LDS (coalesced b32 loads, packed bf16 in LDS), then read as ds_load_u16.
// ACT: 0=none 1=relu 2=sigmoid; compile-time, so the epilogue is branch-free.
// Grid covers M=ROWS exactly -> no bounds checks, EXEC stays all-ones.
// ---------------------------------------------------------------------------
template <int ACT, bool HASBIAS>
__global__ void gemm_wmma_k(const float* __restrict__ A,
                            const float* __restrict__ W,
                            const float* __restrict__ bias,
                            float* __restrict__ out,
                            int N, int K) {
    const int tid  = threadIdx.x;
    const int lane = tid & 31;
    const int wave = tid >> 5;
    const int n0   = blockIdx.x * 16;
    const int m0   = blockIdx.y * 128 + wave * 16;
    const int ln15 = lane & 15;
    const int half = lane >> 4;

    __shared__ __bf16 sB[32][17];    // 32(K) x 16(N), padded

    v8f acc = {};
    for (int k0 = 0; k0 < K; k0 += 32) {
        // ---- cooperative W-tile stage: 512 elems / 256 threads ----------
        __syncthreads();
        {
            int e0 = tid;            // (r,c) = (e>>4, e&15)
            sB[e0 >> 4][e0 & 15] =
                (__bf16)W[(size_t)(k0 + (e0 >> 4)) * N + n0 + (e0 & 15)];
            int e1 = tid + 256;
            sB[e1 >> 4][e1 & 15] =
                (__bf16)W[(size_t)(k0 + (e1 >> 4)) * N + n0 + (e1 & 15)];
        }
        __syncthreads();
        // ---- A fragment: four b128 loads per lane -----------------------
        const float* arow = A + (size_t)(m0 + ln15) * K + k0;
        __builtin_prefetch(arow + 32, 0, 1);         // global_prefetch
        float4 a_lo = *reinterpret_cast<const float4*>(arow + half * 8);
        float4 a_l2 = *reinterpret_cast<const float4*>(arow + half * 8 + 4);
        float4 a_hi = *reinterpret_cast<const float4*>(arow + 16 + half * 8);
        float4 a_h2 = *reinterpret_cast<const float4*>(arow + 16 + half * 8 + 4);
        v16bf a;
        a[0]  = (__bf16)a_lo.x; a[1]  = (__bf16)a_lo.y;
        a[2]  = (__bf16)a_lo.z; a[3]  = (__bf16)a_lo.w;
        a[4]  = (__bf16)a_l2.x; a[5]  = (__bf16)a_l2.y;
        a[6]  = (__bf16)a_l2.z; a[7]  = (__bf16)a_l2.w;
        a[8]  = (__bf16)a_hi.x; a[9]  = (__bf16)a_hi.y;
        a[10] = (__bf16)a_hi.z; a[11] = (__bf16)a_hi.w;
        a[12] = (__bf16)a_h2.x; a[13] = (__bf16)a_h2.y;
        a[14] = (__bf16)a_h2.z; a[15] = (__bf16)a_h2.w;
        // ---- B fragment from LDS: lane = N col, K = table order ----------
        v16bf b;
#pragma unroll
        for (int j = 0; j < 16; ++j) b[j] = sB[(j < 8) ? (half * 8 + j) : (8 + half * 8 + j)][ln15];
        acc = __builtin_amdgcn_wmma_f32_16x16x32_bf16(false, a, false, b,
                                                      (short)0, acc, false, false);
    }
    const float bv = HASBIAS ? bias[n0 + ln15] : 0.0f;
#pragma unroll
    for (int r = 0; r < 8; ++r) {
        float v = acc[r] + bv;
        if (ACT == 1) v = fmaxf(v, 0.0f);
        if (ACT == 2) v = sigm(v);
        out[(size_t)(m0 + r + 8 * half) * N + n0 + ln15] = v;
    }
}

// ---------------------------------------------------------------------------
// Elementwise helpers
// ---------------------------------------------------------------------------
__global__ void add_k(const float* __restrict__ a, const float* __restrict__ b,
                      float* __restrict__ o) {
    int i = blockIdx.x * blockDim.x + threadIdx.x;
    o[i] = a[i] + b[i];
}

// seqs = (item_h * w + user_h * (1 - w)) * sqrt(D)
__global__ void blend_k(const float* __restrict__ ih, const float* __restrict__ uh,
                        const float* __restrict__ w, float* __restrict__ o) {
    int i = blockIdx.x * blockDim.x + threadIdx.x;
    float a = w[i];
    o[i] = (ih[i] * a + uh[i] * (1.0f - a)) * 16.0f;   // sqrt(256)
}

// x += silu(U) * ao   (U = proj[:, 0:256])
__global__ void gated_k(float* __restrict__ x, const float* __restrict__ proj,
                        const float* __restrict__ ao) {
    int i = blockIdx.x * blockDim.x + threadIdx.x;
    int row = i >> 8;
    int col = i & 255;
    float u = proj[(size_t)row * PROJC + col];
    x[i] += (u * sigm(u)) * ao[i];
}

__global__ void resadd_k(float* __restrict__ x, const float* __restrict__ t) {
    int i = blockIdx.x * blockDim.x + threadIdx.x;
    x[i] += t[i];
}

// ---------------------------------------------------------------------------
// LayerNorm: one 256-thread block per row
// ---------------------------------------------------------------------------
__global__ void ln_k(const float* __restrict__ x, const float* __restrict__ g,
                     const float* __restrict__ b, float* __restrict__ out) {
    __shared__ float r1[256];
    __shared__ float r2[256];
    const int row = blockIdx.x;
    const int tid = threadIdx.x;
    float v = x[(size_t)row * DIM + tid];
    r1[tid] = v;
    r2[tid] = v * v;
    __syncthreads();
    for (int s = 128; s > 0; s >>= 1) {
        if (tid < s) { r1[tid] += r1[tid + s]; r2[tid] += r2[tid + s]; }
        __syncthreads();
    }
    float mu  = r1[0] * (1.0f / DIM);
    float var = r2[0] * (1.0f / DIM) - mu * mu;
    out[(size_t)row * DIM + tid] = (v - mu) * rsqrtf(var + 1e-8f) * g[tid] + b[tid];
}

// ---------------------------------------------------------------------------
// RoPE in place on Q (cols 512..767) and K (cols 768..1023) of proj.
// One thread per (row, head, k<16) pair; handles (d=k, d=k+16) rotation.
// ---------------------------------------------------------------------------
__global__ void rope_k(float* __restrict__ proj) {
    int t = blockIdx.x * blockDim.x + threadIdx.x;   // ROWS*HEADS*16 threads
    int k   = t & 15;
    int tmp = t >> 4;
    int h   = tmp & 7;           // HEADS == 8
    int row = tmp >> 3;
    int pos = row & (SEQL - 1);  // L power of two
    // inv_freq[k] = 10000^(-2k/32) = exp(-k * ln(10000)/16)
    float inv = __expf(-(float)k * 0.57564627324851142f);
    float ang = (float)pos * inv;
    float c = __cosf(ang), s = __sinf(ang);
    size_t base = (size_t)row * PROJC + h * HDIM;
    float* Q = proj + base + 2 * DIM;
    float q1 = Q[k], q2 = Q[k + 16];
    Q[k]      = q1 * c - q2 * s;
    Q[k + 16] = q2 * c + q1 * s;
    float* Kp = proj + base + 3 * DIM;
    float k1 = Kp[k], k2 = Kp[k + 16];
    Kp[k]      = k1 * c - k2 * s;
    Kp[k + 16] = k2 * c + k1 * s;
}

// ---------------------------------------------------------------------------
// Sigmoid attention, one wave (32 threads) per (b, h, 16-query tile).
// scores 16x16 via WMMA (K = HD = 32); probs transposed through LDS into an
// A-fragment; V tile staged in LDS (coalesced b128 row loads) for the two PV
// WMMAs. Causal bound trims the key loop.
// ---------------------------------------------------------------------------
__global__ void attn_k(const float* __restrict__ proj,
                       const int* __restrict__ mask,
                       float* __restrict__ ao) {
    const int qt = blockIdx.x;           // L/16
    const int h  = blockIdx.y;           // HEADS
    const int b  = blockIdx.z;           // BATCH
    const int lane = threadIdx.x;
    const int ln15 = lane & 15;
    const int half = lane >> 4;
    const int q0 = qt * 16;
    const int rowbase = b * SEQL;
    const float scale = 0.17677669529663687f;   // HD^-0.5

    // Q fragment (RoPE'd Q at col 512 + h*32), loaded once, b128 loads
    const float* qrow = proj + (size_t)(rowbase + q0 + ln15) * PROJC + 2 * DIM + h * HDIM;
    v16bf aq;
    {
        float4 lo  = *reinterpret_cast<const float4*>(qrow + half * 16);
        float4 lo2 = *reinterpret_cast<const float4*>(qrow + half * 16 + 4);
        float4 hi  = *reinterpret_cast<const float4*>(qrow + half * 16 + 8);
        float4 hi2 = *reinterpret_cast<const float4*>(qrow + half * 16 + 12);
        aq[0]  = (__bf16)lo.x;  aq[1]  = (__bf16)lo.y;
        aq[2]  = (__bf16)lo.z;  aq[3]  = (__bf16)lo.w;
        aq[4]  = (__bf16)lo2.x; aq[5]  = (__bf16)lo2.y;
        aq[6]  = (__bf16)lo2.z; aq[7]  = (__bf16)lo2.w;
        aq[8]  = (__bf16)hi.x;  aq[9]  = (__bf16)hi.y;
        aq[10] = (__bf16)hi.z;  aq[11] = (__bf16)hi.w;
        aq[12] = (__bf16)hi2.x; aq[13] = (__bf16)hi2.y;
        aq[14] = (__bf16)hi2.z; aq[15] = (__bf16)hi2.w;
    }

    __shared__ float sP[16][34];   // prob tile (16q x 32k), padded
    __shared__ float sV[32][33];   // V tile   (32k x 32d), padded
    v8f acc0 = {}, acc1 = {};

    const int nk = (q0 + 16 + 31) >> 5;          // 32-key steps (causal bound)
    for (int kt = 0; kt < nk; ++kt) {
        const int kbase = kt * 32;
        // ---- scores: two 16-key WMMAs (K-frag = 4 b128 loads/lane) ------
        v8f s0, s1;
        {
            const float* krow = proj + (size_t)(rowbase + kbase + ln15) * PROJC + 3 * DIM + h * HDIM;
            float4 k0v = *reinterpret_cast<const float4*>(krow + half * 16);
            float4 k1v = *reinterpret_cast<const float4*>(krow + half * 16 + 4);
            float4 k2v = *reinterpret_cast<const float4*>(krow + half * 16 + 8);
            float4 k3v = *reinterpret_cast<const float4*>(krow + half * 16 + 12);
            v16bf bk;
            bk[0]  = (__bf16)k0v.x; bk[1]  = (__bf16)k0v.y;
            bk[2]  = (__bf16)k0v.z; bk[3]  = (__bf16)k0v.w;
            bk[4]  = (__bf16)k1v.x; bk[5]  = (__bf16)k1v.y;
            bk[6]  = (__bf16)k1v.z; bk[7]  = (__bf16)k1v.w;
            bk[8]  = (__bf16)k2v.x; bk[9]  = (__bf16)k2v.y;
            bk[10] = (__bf16)k2v.z; bk[11] = (__bf16)k2v.w;
            bk[12] = (__bf16)k3v.x; bk[13] = (__bf16)k3v.y;
            bk[14] = (__bf16)k3v.z; bk[15] = (__bf16)k3v.w;
            v8f z = {};
            s0 = __builtin_amdgcn_wmma_f32_16x16x32_bf16(false, aq, false, bk,
                                                         (short)0, z, false, false);
        }
        {
            const float* krow = proj + (size_t)(rowbase + kbase + 16 + ln15) * PROJC + 3 * DIM + h * HDIM;
            float4 k0v = *reinterpret_cast<const float4*>(krow + half * 16);
            float4 k1v = *reinterpret_cast<const float4*>(krow + half * 16 + 4);
            float4 k2v = *reinterpret_cast<const float4*>(krow + half * 16 + 8);
            float4 k3v = *reinterpret_cast<const float4*>(krow + half * 16 + 12);
            v16bf bk;
            bk[0]  = (__bf16)k0v.x; bk[1]  = (__bf16)k0v.y;
            bk[2]  = (__bf16)k0v.z; bk[3]  = (__bf16)k0v.w;
            bk[4]  = (__bf16)k1v.x; bk[5]  = (__bf16)k1v.y;
            bk[6]  = (__bf16)k1v.z; bk[7]  = (__bf16)k1v.w;
            bk[8]  = (__bf16)k2v.x; bk[9]  = (__bf16)k2v.y;
            bk[10] = (__bf16)k2v.z; bk[11] = (__bf16)k2v.w;
            bk[12] = (__bf16)k3v.x; bk[13] = (__bf16)k3v.y;
            bk[14] = (__bf16)k3v.z; bk[15] = (__bf16)k3v.w;
            v8f z = {};
            s1 = __builtin_amdgcn_wmma_f32_16x16x32_bf16(false, aq, false, bk,
                                                         (short)0, z, false, false);
        }
        __syncthreads();   // protect sP/sV against previous iteration reads
        // ---- mask + sigmoid, stage prob tile in LDS ---------------------
        {
            int key = kbase + ln15;
            int pm  = mask[b * SEQL + key];
#pragma unroll
            for (int r = 0; r < 8; ++r) {
                int q = q0 + r + 8 * half;
                sP[r + 8 * half][ln15] =
                    (pm != 0 && key <= q) ? sigm(s0[r] * scale) : 0.0f;
            }
            key = kbase + 16 + ln15;
            pm  = mask[b * SEQL + key];
#pragma unroll
            for (int r = 0; r < 8; ++r) {
                int q = q0 + r + 8 * half;
                sP[r + 8 * half][16 + ln15] =
                    (pm != 0 && key <= q) ? sigm(s1[r] * scale) : 0.0f;
            }
        }
        // ---- stage V tile: lane -> one key row, 8x b128 coalesced -------
        {
            const float* vrow = proj + (size_t)(rowbase + kbase + lane) * PROJC + DIM + h * HDIM;
#pragma unroll
            for (int j = 0; j < 8; ++j) {
                float4 v4 = *reinterpret_cast<const float4*>(vrow + j * 4);
                sV[lane][j * 4 + 0] = v4.x;
                sV[lane][j * 4 + 1] = v4.y;
                sV[lane][j * 4 + 2] = v4.z;
                sV[lane][j * 4 + 3] = v4.w;
            }
        }
        __syncthreads();
        // ---- prob A-fragment (16 queries x 32 keys) ---------------------
        v16bf ap;
#pragma unroll
        for (int j = 0; j < 8; ++j) ap[j]     = (__bf16)sP[ln15][half * 16 + j];
#pragma unroll
        for (int j = 0; j < 8; ++j) ap[8 + j] = (__bf16)sP[ln15][half * 16 + 8 + j];
        // ---- PV: two WMMAs over the 32-dim head, B-frags from LDS -------
        {
            v16bf bv;
#pragma unroll
            for (int j = 0; j < 16; ++j) bv[j] = (__bf16)sV[half * 16 + j][ln15];
            acc0 = __builtin_amdgcn_wmma_f32_16x16x32_bf16(false, ap, false, bv,
                                                           (short)0, acc0, false, false);
        }
        {
            v16bf bv;
#pragma unroll
            for (int j = 0; j < 16; ++j) bv[j] = (__bf16)sV[half * 16 + j][16 + ln15];
            acc1 = __builtin_amdgcn_wmma_f32_16x16x32_bf16(false, ap, false, bv,
                                                           (short)0, acc1, false, false);
        }
    }
    // ---- store attention output [q, h*32 + d] ---------------------------
#pragma unroll
    for (int r = 0; r < 8; ++r) {
        size_t orow = (size_t)(rowbase + q0 + r + 8 * half) * DIM + h * HDIM;
        ao[orow + ln15]      = acc0[r];
        ao[orow + 16 + ln15] = acc1[r];
    }
}

// ---------------------------------------------------------------------------
// Driver
// ---------------------------------------------------------------------------
extern "C" void kernel_launch(void* const* d_in, const int* in_sizes, int n_in,
                              void* d_out, int out_size, void* d_ws, size_t ws_size,
                              hipStream_t stream) {
    const int*   log_seqs   = (const int*)  d_in[0];
    const int*   mask       = (const int*)  d_in[1];
    const float* item_table = (const float*)d_in[2];
    const float* user_table = (const float*)d_in[3];
    const float* itemdnn_w  = (const float*)d_in[4];
    const float* itemdnn_b  = (const float*)d_in[5];
    const float* userdnn_w  = (const float*)d_in[6];
    const float* userdnn_b  = (const float*)d_in[7];
    const float* se_w1      = (const float*)d_in[8];
    const float* se_w2      = (const float*)d_in[9];
    const float* ln1_g      = (const float*)d_in[10];
    const float* ln1_b      = (const float*)d_in[11];
    const float* proj_w     = (const float*)d_in[12];
    const float* proj_b     = (const float*)d_in[13];
    const float* out_w      = (const float*)d_in[14];
    const float* out_b      = (const float*)d_in[15];
    const float* ln2_g      = (const float*)d_in[16];
    const float* ln2_b      = (const float*)d_in[17];
    const float* last_g     = (const float*)d_in[18];
    const float* last_b     = (const float*)d_in[19];
    float* out = (float*)d_out;

    // Workspace carve-up (reused 4 MB chunks + proj)
    const size_t C = (size_t)ROWS * DIM;     // 1,048,576 floats
    float* ws = (float*)d_ws;
    float* c0   = ws + 0 * C;   // item_emb -> combined -> attw
    float* c1   = ws + 1 * C;   // user_emb -> se_hidden
    float* ih   = ws + 2 * C;   // item_h
    float* uh   = ws + 3 * C;   // user_h
    float* x    = ws + 4 * C;   // residual stream
    float* xn   = ws + 5 * C;   // normed
    float* ao   = ws + 6 * C;   // attention output
    float* tmp  = ws + 7 * C;   // out-proj result
    float* proj = ws + 8 * C;   // ROWS x 1024

    const dim3 b256(256);
    const dim3 gElem((unsigned)(C / 256));          // 4096 blocks for ROWS*DIM
    const dim3 gLN(ROWS);

    // 1) embedding gather
    gather_k<<<gElem, b256, 0, stream>>>(log_seqs, mask, item_table, user_table, c0, c1);

    // 2) item/user DNN (relu, bias)
    gemm_wmma_k<1, true><<<dim3(DIM / 16, ROWS / 128), b256, 0, stream>>>(
        c0, itemdnn_w, itemdnn_b, ih, DIM, DIM);
    gemm_wmma_k<1, true><<<dim3(DIM / 16, ROWS / 128), b256, 0, stream>>>(
        c1, userdnn_w, userdnn_b, uh, DIM, DIM);

    // 3) SENet fusion
    add_k<<<gElem, b256, 0, stream>>>(ih, uh, c0);                              // combined
    gemm_wmma_k<1, false><<<dim3(64 / 16, ROWS / 128), b256, 0, stream>>>(
        c0, se_w1, nullptr, c1, 64, DIM);                                       // relu
    gemm_wmma_k<2, false><<<dim3(DIM / 16, ROWS / 128), b256, 0, stream>>>(
        c1, se_w2, nullptr, c0, DIM, 64);                                       // sigmoid
    blend_k<<<gElem, b256, 0, stream>>>(ih, uh, c0, x);                         // * sqrt(D)

    // 4) transformer blocks
    for (int i = 0; i < NBLK; ++i) {
        ln_k<<<gLN, b256, 0, stream>>>(x, ln1_g + i * DIM, ln1_b + i * DIM, xn);
        gemm_wmma_k<0, true><<<dim3(PROJC / 16, ROWS / 128), b256, 0, stream>>>(
            xn, proj_w + (size_t)i * DIM * PROJC, proj_b + i * PROJC,
            proj, PROJC, DIM);
        rope_k<<<dim3(ROWS * HEADS * 16 / 256), b256, 0, stream>>>(proj);
        attn_k<<<dim3(SEQL / 16, HEADS, BATCH), dim3(32), 0, stream>>>(proj, mask, ao);
        gated_k<<<gElem, b256, 0, stream>>>(x, proj, ao);
        ln_k<<<gLN, b256, 0, stream>>>(x, ln2_g + i * DIM, ln2_b + i * DIM, xn);
        gemm_wmma_k<0, true><<<dim3(DIM / 16, ROWS / 128), b256, 0, stream>>>(
            xn, out_w + (size_t)i * DIM * DIM, out_b + i * DIM,
            tmp, DIM, DIM);
        resadd_k<<<gElem, b256, 0, stream>>>(x, tmp);
    }

    // 5) final layernorm -> output
    ln_k<<<gLN, b256, 0, stream>>>(x, last_g, last_b, out);
}
